// EdgeNetwork_78091095376039
// MI455X (gfx1250) — compile-verified, matrix operand order checked by reading, not run on hardware
//
#include <hip/hip_runtime.h>

typedef _Float16 v16h __attribute__((ext_vector_type(16)));
typedef float    v8f  __attribute__((ext_vector_type(8)));
typedef _Float16 h2   __attribute__((ext_vector_type(2)));

#define N_ATOMS  100000
#define N_EDGES  800000
#define ATOM_DIM 32
#define BOND_DIM 16
#define NTILES   (N_EDGES / 16)     // 50000 tiles of 16 edges, exact
#define KSTEPS   17                 // 16 bond-k steps + 1 bias step
#define WL_DW    (KSTEPS * 2 * 32 * 8)  // W fragments: 17 steps x 2 n-tiles x 32 lanes x 8 dwords

static __device__ __forceinline__ h2 mkh2(float a, float b) {
    h2 r; r[0] = (_Float16)a; r[1] = (_Float16)b; return r;
}

union AB { h2 h[8]; v16h v; unsigned u[8]; };

__global__ void zero_f32(float* __restrict__ p, int n) {
    int i = blockIdx.x * blockDim.x + threadIdx.x;
    if (i < n) p[i] = 0.0f;
}

__global__ __launch_bounds__(256)
void edge_net(const float* __restrict__ atom,   // (N_ATOMS, 32)
              const float* __restrict__ bond,   // (N_EDGES, 16)
              const int*   __restrict__ pairs,  // (N_EDGES, 2)  [src, dst]
              const float* __restrict__ kern,   // (16, 1024)
              const float* __restrict__ bias,   // (1024,)
              float*       __restrict__ out) {  // (N_ATOMS, 32), pre-zeroed
    // -------- Stage 0: pack W (544 x 32, f16) into LDS in WMMA-B fragment order.
    // B frag for (step t, n-tile nt): lane l covers column i = nt*16 + (l&15);
    // dword d holds K-halves c, c+1 with c = bc + (d<4 ? 2d : 16+2(d-4)), bc = (l>>4)*8.
    // Global K index kappa = 32*t + c:  t<16 -> kernel[t, i*32+c];  t==16 -> bias[i*32+c].
    __shared__ __align__(32) unsigned WL[WL_DW];
    const int tid = threadIdx.x;
    for (int q = tid; q < WL_DW; q += 256) {
        int d    = q & 7;
        int lane = (q >> 3) & 31;
        int nt   = (q >> 8) & 1;
        int t    = q >> 9;
        int i    = nt * 16 + (lane & 15);
        int bc   = (lane >> 4) * 8;
        int c    = bc + ((d < 4) ? (2 * d) : (16 + 2 * (d - 4)));
        float f0, f1;
        if (t < BOND_DIM) {
            const float* kp = kern + (size_t)t * (ATOM_DIM * ATOM_DIM) + i * ATOM_DIM;
            f0 = kp[c]; f1 = kp[c + 1];
        } else {
            const float* bp = bias + i * ATOM_DIM;
            f0 = bp[c]; f1 = bp[c + 1];
        }
        AB tmp; tmp.h[0] = mkh2(f0, f1);
        WL[q] = tmp.u[0];
    }
    __syncthreads();
    const v16h* __restrict__ WV = (const v16h*)WL;

    const int lane = tid & 31;
    const int wv   = tid >> 5;
    const int M    = lane & 15;      // A row / C column within tile (lanes 0-15 and 16-31 alias M)
    const int half = lane >> 4;      // K-half selector for A fragment
    const int bc   = half * 8;
    const int nwaves = gridDim.x * 8;

    // -------- Hoist ALL 34 B fragments into registers (tile-invariant, 272 VGPRs).
    // Removes every per-step DS load from the hot loop.
    v16h Breg[KSTEPS * 2];
#pragma unroll
    for (int t = 0; t < KSTEPS; ++t) {
        Breg[2 * t + 0] = WV[(t * 2 + 0) * 32 + lane];
        Breg[2 * t + 1] = WV[(t * 2 + 1) * 32 + lane];
    }

    for (int tile = blockIdx.x * 8 + wv; tile < NTILES; tile += nwaves) {
        const int e0 = tile * 16;
        const int e  = e0 + M;
        const int src = pairs[2 * e];
        const int dst = pairs[2 * e + 1];

        // -------- Neigh fragment (16 halves covering this lane's K-set) in registers.
        const float* ar = atom + (size_t)dst * ATOM_DIM + bc;
        float4 c0 = ((const float4*)ar)[0];
        float4 c1 = ((const float4*)ar)[1];
        float4 c2 = ((const float4*)(ar + 16))[0];
        float4 c3 = ((const float4*)(ar + 16))[1];
        AB nf;
        nf.h[0] = mkh2(c0.x, c0.y);  nf.h[1] = mkh2(c0.z, c0.w);
        nf.h[2] = mkh2(c1.x, c1.y);  nf.h[3] = mkh2(c1.z, c1.w);
        nf.h[4] = mkh2(c2.x, c2.y);  nf.h[5] = mkh2(c2.z, c2.w);
        nf.h[6] = mkh2(c3.x, c3.y);  nf.h[7] = mkh2(c3.z, c3.w);

        // Bond row; compiler broadcasts single f16 halves via op_sel on v_pk_mul_f16.
        const float* br = bond + (size_t)e * BOND_DIM;
        float4 b0 = ((const float4*)br)[0];
        float4 b1 = ((const float4*)br)[1];
        float4 b2 = ((const float4*)br)[2];
        float4 b3 = ((const float4*)br)[3];
        const float bb[16] = { b0.x, b0.y, b0.z, b0.w,
                               b1.x, b1.y, b1.z, b1.w,
                               b2.x, b2.y, b2.z, b2.w,
                               b3.x, b3.y, b3.z, b3.w };

        // A fragments. Each af[t] is consumed both by its WMMA and by the liveness
        // pin below, so RA must keep 16 distinct register sets: no WMMA source reg
        // is rewritten inside the chain -> no WAR hazard v_nops.
        AB af[BOND_DIM];
#pragma unroll
        for (int t = 0; t < BOND_DIM; ++t) {
            h2 bs = mkh2(bb[t], bb[t]);
#pragma unroll
            for (int d = 0; d < 8; ++d) af[t].h[d] = bs * nf.h[d];  // v_pk_mul_f16
        }

        v8f acc0 = {};   // columns i = 0..15
        v8f acc1 = {};   // columns i = 16..31

#pragma unroll
        for (int t = 0; t < BOND_DIM; ++t) {
            acc0 = __builtin_amdgcn_wmma_f32_16x16x32_f16(false, af[t].v, false, Breg[2 * t + 0],
                                                          (short)0, acc0, false, false);
            acc1 = __builtin_amdgcn_wmma_f32_16x16x32_f16(false, af[t].v, false, Breg[2 * t + 1],
                                                          (short)0, acc1, false, false);
        }
        // Bias step (weight 1): A fragment is the raw neigh fragment.
        acc0 = __builtin_amdgcn_wmma_f32_16x16x32_f16(false, nf.v, false, Breg[32],
                                                      (short)0, acc0, false, false);
        acc1 = __builtin_amdgcn_wmma_f32_16x16x32_f16(false, nf.v, false, Breg[33],
                                                      (short)0, acc1, false, false);

        // Liveness pin: keeps all 16 A fragments live across the WMMA chain so the
        // register allocator cannot coalesce them into one buffer. Emits nothing.
        asm volatile("" ::
            "v"(af[0].v),  "v"(af[1].v),  "v"(af[2].v),  "v"(af[3].v),
            "v"(af[4].v),  "v"(af[5].v),  "v"(af[6].v),  "v"(af[7].v),
            "v"(af[8].v),  "v"(af[9].v),  "v"(af[10].v), "v"(af[11].v),
            "v"(af[12].v), "v"(af[13].v), "v"(af[14].v), "v"(af[15].v));

        // -------- Epilogue: scatter-add into out[src]. C layout: VGPR r, lane l ->
        // row Mr = r + 8*(l>>4), col N = l&15 (acc0: i=N, acc1: i=N+16).
#pragma unroll
        for (int r = 0; r < 8; ++r) {
            int Mr = r + 8 * half;
            int s  = __shfl(src, Mr, 32);   // src index of edge e0+Mr, held by lane Mr
            atomicAdd(&out[(size_t)s * ATOM_DIM + M],       acc0[r]);
            atomicAdd(&out[(size_t)s * ATOM_DIM + 16 + M],  acc1[r]);
        }
    }
}

extern "C" void kernel_launch(void* const* d_in, const int* in_sizes, int n_in,
                              void* d_out, int out_size, void* d_ws, size_t ws_size,
                              hipStream_t stream) {
    const float* atom = (const float*)d_in[0];
    const float* bond = (const float*)d_in[1];
    const int*   pair = (const int*)  d_in[2];
    const float* kern = (const float*)d_in[3];
    const float* bias = (const float*)d_in[4];
    float* out = (float*)d_out;

    const int n_out = N_ATOMS * ATOM_DIM;
    zero_f32<<<(n_out + 255) / 256, 256, 0, stream>>>(out, n_out);

    // 256 threads = 8 waves per WG; waves grid-stride over 50000 16-edge tiles.
    edge_net<<<1024, 256, 0, stream>>>(atom, bond, pair, kern, bias, out);
}
